// xLSTM_Model2_11312943858296
// MI455X (gfx1250) — matrix operand order repaired and checked by reading, not compiled, not run on hardware
//
#include <hip/hip_runtime.h>
#include <hip/hip_bf16.h>
#include <stdint.h>

// ---------------------------------------------------------------------------
// xLSTM forward for MI455X (gfx1250, wave32).
//
//  * GEMMs on v_wmma_f32_16x16x32_bf16 (fp32 accumulate), 4 N-tiles per wave
//    (A-fragment reused 4x, 32 WMMAs per block per K-step).
//  * B panels (4KB, kt-major contiguous) are DMA'd into LDS by the Tensor
//    Data Mover (tensor_load_to_lds) with double buffering, synchronized via
//    s_wait_tensorcnt + workgroup barrier.
//  * Per (t, layer): LN+pack -> gates GEMM (fused sigmoid/tanh) -> attention
//    gate -> cell update+LN (packs h) -> residual MLP (two GEMMs, relu
//    epilogue packs A-fragments directly) -> residual LN.
//  * Embed/skip (K=17) precomputed for all T; output head is one batched
//    WMMA GEMM over T*B = 32768 rows.
// ---------------------------------------------------------------------------

#define HID    256
#define H2     128
#define NFEAT  16
#define TFEAT  17
#define BATCH  128
#define TSTEPS 256
#define NLAYER 3
#define LN_EPS 1e-5f

typedef __bf16 bf16_t;
typedef bf16_t v16bf __attribute__((ext_vector_type(16)));
typedef float  v8f   __attribute__((ext_vector_type(8)));
typedef unsigned int u32x4 __attribute__((ext_vector_type(4)));
typedef int i32x8 __attribute__((ext_vector_type(8)));
typedef int i32x4 __attribute__((ext_vector_type(4)));

enum { MODE_NONE = 0, MODE_GATES = 1, MODE_RELU_PACK = 2 };

// ------------------------------ device helpers ------------------------------

__device__ __forceinline__ float block_sum(float v, float* red) {
  const int t = threadIdx.x;
  red[t] = v;
  __syncthreads();
  for (int off = blockDim.x >> 1; off > 0; off >>= 1) {
    if (t < off) red[t] += red[t + off];
    __syncthreads();
  }
  float r = red[0];
  __syncthreads();
  return r;
}

__device__ __forceinline__ float sigmoidf_(float x) {
  return 1.0f / (1.0f + __expf(-x));
}

// Pack one scalar of an M x K activation matrix into the WMMA A-fragment
// tile layout for 16x16x32 bf16 (ISA 7.12.2 "16-bit A-Matrix 16x32").
__device__ __forceinline__ void pack_a_elem(bf16_t* dst, int Ktiles,
                                            int m, int k, float val) {
  const int kt   = k >> 5;
  const int kk   = k & 31;
  const int lane = (m & 15) + ((kk & 8) ? 16 : 0);
  const int j    = (kk & 16) ? ((kk & 7) + 8) : (kk & 7);
  dst[(((size_t)(m >> 4)) * Ktiles + kt) * 512 + lane * 16 + j] = (bf16_t)val;
}

// Tensor Data Mover: 1-D DMA of `nelem` 2-byte elements from global to LDS.
// D# per CDNA5 ISA: group0 {count=1, lds_addr,
// global_addr, type=2}; group1 {data_size=1(2B), tensor_dim0, tensor_dim1=1,
// tile_dim0, stride}; groups 2/3 zero (<=2D tensor).
// Toolchain here is the 6-arg clang-23 form:
//   (u32x4 g0, i32x8 g1, i32x4 g2, i32x4 g3, i32x8 pad, i32 cpol)
__device__ __forceinline__ void tdm_load_1d(void* lds_dst, const void* gsrc,
                                            int nelem) {
  const unsigned lds = (unsigned)(size_t)lds_dst;
  const unsigned long long ga = (unsigned long long)(size_t)gsrc;
  u32x4 g0;
  g0[0] = 1u;                                              // count=1 (valid)
  g0[1] = lds;                                             // lds_addr
  g0[2] = (unsigned)(ga & 0xFFFFFFFFu);                    // global_addr lo
  g0[3] = (unsigned)((ga >> 32) & 0x01FFFFFFu) | (2u << 30); // addr hi | type=2
  i32x8 g1;
  g1[0] = (1 << 16);                                       // data_size = 2B
  g1[1] = (int)(((unsigned)nelem & 0xFFFFu) << 16);        // tensor_dim0 lo16
  g1[2] = (int)(((unsigned)nelem >> 16) | (1u << 16));     // dim0 hi | dim1=1
  g1[3] = (int)(((unsigned)nelem & 0xFFFFu) << 16);        // tile_dim0
  g1[4] = 0;                                               // tile_dim1/2 unused
  g1[5] = nelem;                                           // dim0 stride lo32
  g1[6] = 0;
  g1[7] = 0;
  i32x4 z4 = {0, 0, 0, 0};
  i32x8 z8 = {0, 0, 0, 0, 0, 0, 0, 0};
  __builtin_amdgcn_tensor_load_to_lds(g0, g1, z4, z4, z8, 0);
}

// ------------------------------ weight packing ------------------------------

// Pack a K x N fp32 weight (row-major) into bf16 WMMA B-fragment tiles,
// kt-major: tile (kt, nt) so a fixed-kt panel of consecutive nt is contiguous.
__global__ void k_packw(const float* __restrict__ src, bf16_t* __restrict__ dst,
                        int K, int N) {
  const int idx = blockIdx.x * 256 + threadIdx.x;
  if (idx >= K * N) return;
  const int k = idx / N, n = idx % N;
  const int kt = k >> 5, kk = k & 31, nt = n >> 4, nn = n & 15;
  const int lane = nn + ((kk & 8) ? 16 : 0);
  const int j = (kk & 16) ? ((kk & 7) + 8) : (kk & 7);
  const int Ntiles = N >> 4;
  dst[(((size_t)kt) * Ntiles + nt) * 512 + lane * 16 + j] = (bf16_t)src[idx];
}

// ------------------------------ WMMA GEMM -----------------------------------
// C[M,N] = A[M,K]@B[K,N] + bias with fused epilogue. Block = 8 waves; wave w
// owns mtile = blockIdx.y*8 + w and 4 ntiles (blockIdx.x*4 .. +3). The 4KB B
// panel per K-step is TDM-loaded into double-buffered LDS by wave 0.
__global__ void __launch_bounds__(256)
k_gemm(const bf16_t* __restrict__ Apk, const bf16_t* __restrict__ Bpk,
       const float* __restrict__ bias, float* __restrict__ Cout,
       bf16_t* __restrict__ CpkOut, int Ktiles, int N, int mode, int dstKt) {
  __shared__ __align__(128) bf16_t Bs[2][4 * 512];
  const int tid    = threadIdx.x;
  const int wave   = tid >> 5;
  const int lane   = tid & 31;
  const int ntBase = blockIdx.x * 4;
  const int mt     = blockIdx.y * 8 + wave;
  const int Ntiles = N >> 4;

  const bf16_t* Abase = Apk + ((size_t)mt * Ktiles) * 512;

  if (tid < 32)  // wave 0 drives the TDM pipeline
    tdm_load_1d(Bs[0], Bpk + ((size_t)0 * Ntiles + ntBase) * 512, 4 * 512);

  v8f acc[4] = {};
  for (int kt = 0; kt < Ktiles; ++kt) {
    if (tid < 32) {
      if (kt + 1 < Ktiles) {
        tdm_load_1d(Bs[(kt + 1) & 1],
                    Bpk + ((size_t)(kt + 1) * Ntiles + ntBase) * 512, 4 * 512);
        __builtin_amdgcn_s_wait_tensorcnt(1);  // panel kt has landed
      } else {
        __builtin_amdgcn_s_wait_tensorcnt(0);
      }
    }
    __syncthreads();  // publish panel kt to all waves

    if (kt + 1 < Ktiles)
      __builtin_prefetch(Abase + (size_t)(kt + 1) * 512, 0, 1);
    v16bf a = *(const v16bf*)(Abase + (size_t)kt * 512 + lane * 16);
    const bf16_t* bp = Bs[kt & 1];
#pragma unroll
    for (int j = 0; j < 4; ++j) {
      v16bf b = *(const v16bf*)(bp + j * 512 + lane * 16);
      acc[j] = __builtin_amdgcn_wmma_f32_16x16x32_bf16(
          false, a, false, b, (short)0, acc[j], false, false);
    }
    __syncthreads();  // all reads of panel kt done before it is overwritten
  }

  const int nn  = lane & 15;
  const int mhi = (lane < 16) ? 0 : 8;
#pragma unroll
  for (int j = 0; j < 4; ++j) {
    const int n = (ntBase + j) * 16 + nn;
    const float bv = bias ? bias[n] : 0.0f;
#pragma unroll
    for (int v = 0; v < 8; ++v) {
      const int m = mt * 16 + v + mhi;
      float x = acc[j][v] + bv;
      if (mode == MODE_GATES) {
        const int q = n >> 8;  // 0:i 1:f 2:g 3:o over N=1024
        x = (q == 2) ? tanhf(x) : sigmoidf_(x);
        Cout[(size_t)m * N + n] = x;
      } else if (mode == MODE_RELU_PACK) {
        x = fmaxf(x, 0.0f);
        pack_a_elem(CpkOut, dstKt, m, n, x);
      } else {
        Cout[(size_t)m * N + n] = x;
      }
    }
  }
}

// ------------------------- embed / skip precompute --------------------------

__global__ void __launch_bounds__(256)
k_embed(const float* __restrict__ params, const float* __restrict__ disp,
        const float* __restrict__ ew, const float* __restrict__ eb,
        const float* __restrict__ eg, const float* __restrict__ ebeta,
        float* __restrict__ x0all) {
  const int r = blockIdx.x;            // r = t*B + b
  const int t = r / BATCH, b = r % BATCH;
  const int n = threadIdx.x;
  __shared__ float feat[TFEAT];
  __shared__ float red[HID];
  if (n == 0) feat[0] = disp[(size_t)b * TSTEPS + t];
  else if (n < TFEAT) feat[n] = params[b * NFEAT + n - 1];
  __syncthreads();
  float s = eb[n];
#pragma unroll
  for (int k = 0; k < TFEAT; ++k) s += feat[k] * ew[k * HID + n];
  const float mu  = block_sum(s, red) * (1.0f / HID);
  const float d   = s - mu;
  const float var = block_sum(d * d, red) * (1.0f / HID);
  float v = d * rsqrtf(var + LN_EPS) * eg[n] + ebeta[n];
  x0all[(size_t)r * HID + n] = fmaxf(v, 0.0f);
}

__global__ void __launch_bounds__(128)
k_skip(const float* __restrict__ params, const float* __restrict__ disp,
       const float* __restrict__ sw, const float* __restrict__ sb,
       float* __restrict__ skipall) {
  const int r = blockIdx.x;
  const int t = r / BATCH, b = r % BATCH;
  const int n = threadIdx.x;
  __shared__ float feat[TFEAT];
  if (n == 0) feat[0] = disp[(size_t)b * TSTEPS + t];
  else if (n < TFEAT) feat[n] = params[b * NFEAT + n - 1];
  __syncthreads();
  float s = sb[n];
#pragma unroll
  for (int k = 0; k < TFEAT; ++k) s += feat[k] * sw[k * H2 + n];
  skipall[(size_t)r * H2 + n] = fmaxf(s, 0.0f);
}

// ----------------------- recurrent per-(t,l) kernels ------------------------

__global__ void __launch_bounds__(256)
k_lnpack(const float* __restrict__ xin, const float* __restrict__ hin,
         const float* __restrict__ ig, const float* __restrict__ ib,
         const float* __restrict__ hg, const float* __restrict__ hb,
         float* __restrict__ xres, bf16_t* __restrict__ Apk) {
  const int b = blockIdx.x, j = threadIdx.x;
  __shared__ float red[HID];
  const float xv = xin[(size_t)b * HID + j];
  const float hv = hin[(size_t)b * HID + j];
  xres[(size_t)b * HID + j] = xv;

  float mu  = block_sum(xv, red) * (1.0f / HID);
  float d   = xv - mu;
  float var = block_sum(d * d, red) * (1.0f / HID);
  const float xn = d * rsqrtf(var + LN_EPS) * ig[j] + ib[j];

  mu  = block_sum(hv, red) * (1.0f / HID);
  d   = hv - mu;
  var = block_sum(d * d, red) * (1.0f / HID);
  const float hn = d * rsqrtf(var + LN_EPS) * hg[j] + hb[j];

  pack_a_elem(Apk, 16, b, j, xn);
  pack_a_elem(Apk, 16, b, HID + j, hn);
}

__global__ void __launch_bounds__(256)
k_attn(const float* __restrict__ c, const float* __restrict__ aw1,
       const float* __restrict__ ab1, const float* __restrict__ aw2,
       const float* __restrict__ ab2, float* __restrict__ imp) {
  const int b = blockIdx.x, k = threadIdx.x;
  __shared__ float cs[HID];
  __shared__ float red[HID];
  cs[k] = c[(size_t)b * HID + k];
  __syncthreads();
  float s = ab1[k];
  for (int d = 0; d < HID; ++d) s += cs[d] * aw1[d * HID + k];
  const float a = tanhf(s) * aw2[k];
  const float tot = block_sum(a, red);
  if (k == 0) imp[b] = sigmoidf_(tot + ab2[0]);
}

__global__ void __launch_bounds__(256)
k_cell(const float* __restrict__ g4, const float* __restrict__ imp,
       float* __restrict__ c, float* __restrict__ h,
       const float* __restrict__ sg, const float* __restrict__ sb,
       bf16_t* __restrict__ ApkR1) {
  const int b = blockIdx.x, j = threadIdx.x;
  __shared__ float red[HID];
  const float iv = g4[(size_t)b * 1024 + j];
  const float fv = g4[(size_t)b * 1024 + 256 + j];
  const float gv = g4[(size_t)b * 1024 + 512 + j];
  const float ov = g4[(size_t)b * 1024 + 768 + j];
  const float cn = (fv * c[(size_t)b * HID + j] + iv * gv) * imp[b];
  const float mu  = block_sum(cn, red) * (1.0f / HID);
  const float d   = cn - mu;
  const float var = block_sum(d * d, red) * (1.0f / HID);
  const float cl  = d * rsqrtf(var + LN_EPS) * sg[j] + sb[j];
  c[(size_t)b * HID + j] = cl;
  const float hv = ov * tanhf(cl);
  h[(size_t)b * HID + j] = hv;
  pack_a_elem(ApkR1, 8, b, j, hv);
}

__global__ void __launch_bounds__(256)
k_residln(const float* __restrict__ rb, const float* __restrict__ h,
          const float* __restrict__ xres, const float* __restrict__ g,
          const float* __restrict__ bb, float* __restrict__ X,
          float* __restrict__ xfin) {
  const int b = blockIdx.x, j = threadIdx.x;
  __shared__ float red[HID];
  const float t = rb[(size_t)b * HID + j] + h[(size_t)b * HID + j];
  const float mu  = block_sum(t, red) * (1.0f / HID);
  const float d   = t - mu;
  const float var = block_sum(d * d, red) * (1.0f / HID);
  const float out = d * rsqrtf(var + LN_EPS) * g[j] + bb[j] +
                    xres[(size_t)b * HID + j];
  X[(size_t)b * HID + j] = out;
  if (xfin) xfin[(size_t)b * HID + j] = out;
}

// ------------------------------- output head --------------------------------

__global__ void __launch_bounds__(256)
k_packy(const float* __restrict__ xf, const float* __restrict__ sk,
        bf16_t* __restrict__ Apk) {
  const long idx = (long)blockIdx.x * 256 + threadIdx.x;
  if (idx >= (long)TSTEPS * BATCH * (HID + H2)) return;
  const int r = (int)(idx / (HID + H2));
  const int k = (int)(idx % (HID + H2));
  const float v = (k < HID) ? xf[(size_t)r * HID + k]
                            : sk[(size_t)r * H2 + (k - HID)];
  pack_a_elem(Apk, 12, r, k, v);
}

__global__ void __launch_bounds__(128)
k_outhead(const float* __restrict__ o1, const float* __restrict__ g,
          const float* __restrict__ bb, const float* __restrict__ w2,
          const float* __restrict__ b2, float* __restrict__ out) {
  const int r = blockIdx.x;          // r = t*B + b
  const int n = threadIdx.x;         // 128
  __shared__ float red[H2];
  const float v = o1[(size_t)r * H2 + n];
  const float mu  = block_sum(v, red) * (1.0f / H2);
  const float d   = v - mu;
  const float var = block_sum(d * d, red) * (1.0f / H2);
  const float ln  = d * rsqrtf(var + LN_EPS) * g[n] + bb[n];
  const float s   = block_sum(fmaxf(ln, 0.0f) * w2[n], red);
  if (n == 0) {
    const int t = r / BATCH, b = r % BATCH;
    out[(size_t)b * TSTEPS + t] = s + b2[0];
  }
}

// ------------------------------- host driver --------------------------------

extern "C" void kernel_launch(void* const* d_in, const int* in_sizes, int n_in,
                              void* d_out, int out_size, void* d_ws, size_t ws_size,
                              hipStream_t stream) {
  (void)in_sizes; (void)n_in; (void)out_size; (void)ws_size;
  const float* params  = (const float*)d_in[0];
  const float* disp    = (const float*)d_in[1];
  const float* embed_w = (const float*)d_in[2];
  const float* embed_b = (const float*)d_in[3];
  const float* embed_g = (const float*)d_in[4];
  const float* embed_be= (const float*)d_in[5];
  const float* skip_w  = (const float*)d_in[6];
  const float* skip_b  = (const float*)d_in[7];
  const float* gates_w = (const float*)d_in[8];
  const float* gates_b = (const float*)d_in[9];
  const float* inln_g  = (const float*)d_in[10];
  const float* inln_b  = (const float*)d_in[11];
  const float* hln_g   = (const float*)d_in[12];
  const float* hln_b   = (const float*)d_in[13];
  const float* sln_g   = (const float*)d_in[14];
  const float* sln_b   = (const float*)d_in[15];
  const float* aw1     = (const float*)d_in[16];
  const float* ab1     = (const float*)d_in[17];
  const float* aw2     = (const float*)d_in[18];
  const float* ab2     = (const float*)d_in[19];
  const float* rw1     = (const float*)d_in[20];
  const float* rb1     = (const float*)d_in[21];
  const float* rw2     = (const float*)d_in[22];
  const float* rb2     = (const float*)d_in[23];
  const float* rln_g   = (const float*)d_in[24];
  const float* rln_b   = (const float*)d_in[25];
  const float* ow1     = (const float*)d_in[26];
  const float* ob1     = (const float*)d_in[27];
  const float* oln_g   = (const float*)d_in[28];
  const float* oln_b   = (const float*)d_in[29];
  const float* ow2     = (const float*)d_in[30];
  const float* ob2     = (const float*)d_in[31];
  float* out = (float*)d_out;

  // ---- workspace carve-up (~99 MB) ----
  char* ws = (char*)d_ws;
  size_t off = 0;
  auto carve = [&](size_t bytes) -> char* {
    char* p = ws + off;
    off = (off + bytes + 255) & ~(size_t)255;
    return p;
  };
  const int R = TSTEPS * BATCH;                              // 32768 rows
  bf16_t* gWpk   = (bf16_t*)carve((size_t)NLAYER * 512 * 1024 * 2);
  bf16_t* rw1pk  = (bf16_t*)carve((size_t)NLAYER * 256 * 512 * 2);
  bf16_t* rw2pk  = (bf16_t*)carve((size_t)NLAYER * 512 * 256 * 2);
  bf16_t* ow1pk  = (bf16_t*)carve((size_t)384 * 128 * 2);
  float*  x0all  = (float*)carve((size_t)R * HID * 4);       // reused as x_final
  float*  skipall= (float*)carve((size_t)R * H2 * 4);
  float*  hbuf   = (float*)carve((size_t)NLAYER * BATCH * HID * 4);
  float*  cbuf   = (float*)carve((size_t)NLAYER * BATCH * HID * 4);
  float*  Xbuf   = (float*)carve((size_t)BATCH * HID * 4);
  float*  Xres   = (float*)carve((size_t)BATCH * HID * 4);
  float*  g4     = (float*)carve((size_t)BATCH * 1024 * 4);
  float*  impb   = (float*)carve((size_t)BATCH * 4);
  bf16_t* ApkG   = (bf16_t*)carve((size_t)8 * 16 * 512 * 2);
  bf16_t* ApkR1  = (bf16_t*)carve((size_t)8 * 8 * 512 * 2);
  bf16_t* ApkR2  = (bf16_t*)carve((size_t)8 * 16 * 512 * 2);
  float*  rbout  = (float*)carve((size_t)BATCH * HID * 4);
  bf16_t* ypk    = (bf16_t*)carve((size_t)2048 * 12 * 512 * 2);
  float*  o1buf  = (float*)carve((size_t)R * H2 * 4);

  // ---- one-time: pack weights to bf16 WMMA B fragments (kt-major) ----
  for (int l = 0; l < NLAYER; ++l) {
    k_packw<<<(512 * 1024) / 256, 256, 0, stream>>>(
        gates_w + (size_t)l * 512 * 1024, gWpk + (size_t)l * 512 * 1024, 512, 1024);
    k_packw<<<(256 * 512) / 256, 256, 0, stream>>>(
        rw1 + (size_t)l * 256 * 512, rw1pk + (size_t)l * 256 * 512, 256, 512);
    k_packw<<<(512 * 256) / 256, 256, 0, stream>>>(
        rw2 + (size_t)l * 512 * 256, rw2pk + (size_t)l * 512 * 256, 512, 256);
  }
  k_packw<<<(384 * 128) / 256, 256, 0, stream>>>(ow1, ow1pk, 384, 128);

  // ---- init state, precompute embed / skip for all T ----
  (void)hipMemsetAsync(hbuf, 0, (size_t)NLAYER * BATCH * HID * 4, stream);
  (void)hipMemsetAsync(cbuf, 0, (size_t)NLAYER * BATCH * HID * 4, stream);
  k_embed<<<R, 256, 0, stream>>>(params, disp, embed_w, embed_b, embed_g,
                                 embed_be, x0all);
  k_skip<<<R, 128, 0, stream>>>(params, disp, skip_w, skip_b, skipall);

  // ---- recurrent loop ----
  for (int t = 0; t < TSTEPS; ++t) {
    const float* xin = x0all + (size_t)t * BATCH * HID;
    for (int l = 0; l < NLAYER; ++l) {
      float* hl = hbuf + (size_t)l * BATCH * HID;
      float* cl = cbuf + (size_t)l * BATCH * HID;
      k_lnpack<<<BATCH, 256, 0, stream>>>(
          xin, hl, inln_g + l * HID, inln_b + l * HID,
          hln_g + l * HID, hln_b + l * HID, Xres, ApkG);
      k_gemm<<<dim3(16, 1), 256, 0, stream>>>(                    // 64 ntiles / 4
          ApkG, gWpk + (size_t)l * 512 * 1024, gates_b + l * 1024,
          g4, nullptr, /*Ktiles=*/16, /*N=*/1024, MODE_GATES, 0);
      k_attn<<<BATCH, 256, 0, stream>>>(
          cl, aw1 + (size_t)l * HID * HID, ab1 + l * HID,
          aw2 + l * HID, ab2 + l, impb);
      k_cell<<<BATCH, 256, 0, stream>>>(
          g4, impb, cl, hl, sln_g + l * HID, sln_b + l * HID, ApkR1);
      k_gemm<<<dim3(8, 1), 256, 0, stream>>>(                     // 32 ntiles / 4
          ApkR1, rw1pk + (size_t)l * 256 * 512, rb1 + l * 512,
          nullptr, ApkR2, /*Ktiles=*/8, /*N=*/512, MODE_RELU_PACK, /*dstKt=*/16);
      k_gemm<<<dim3(4, 1), 256, 0, stream>>>(                     // 16 ntiles / 4
          ApkR2, rw2pk + (size_t)l * 512 * 256, rb2 + l * HID,
          rbout, nullptr, /*Ktiles=*/16, /*N=*/256, MODE_NONE, 0);
      const bool last = (l == NLAYER - 1);
      k_residln<<<BATCH, 256, 0, stream>>>(
          rbout, hl, Xres, rln_g + l * HID, rln_b + l * HID, Xbuf,
          last ? (x0all + (size_t)t * BATCH * HID) : nullptr);
      xin = Xbuf;
    }
  }

  // ---- output head over all T*B rows ----
  {
    const long ytot = (long)R * (HID + H2);
    k_packy<<<(unsigned)((ytot + 255) / 256), 256, 0, stream>>>(x0all, skipall, ypk);
    k_gemm<<<dim3(2, 256), 256, 0, stream>>>(                     // 8 ntiles / 4
        ypk, ow1pk, ob1, o1buf, nullptr, /*Ktiles=*/12, /*N=*/128, MODE_NONE, 0);
    k_outhead<<<R, 128, 0, stream>>>(o1buf, oln_g, oln_b, ow2, ob2, out);
  }
}
